// KVMN_67757404061823
// MI455X (gfx1250) — matrix operand order repaired and checked by reading, not compiled, not run on hardware
//
#include <hip/hip_runtime.h>
#include <hip/hip_bf16.h>
#include <math.h>

#define B_ 256
#define T_ 50
#define NITEMS 100000
#define EMB 256
#define KBD 128
#define HID 256
#define OUT_ 256
#define NF 10
#define G3 768 // 3*HID

typedef __bf16 bf16;
typedef bf16 bf16x16 __attribute__((ext_vector_type(16)));
typedef bf16 bf16x8  __attribute__((ext_vector_type(8)));
typedef float f32x8  __attribute__((ext_vector_type(8)));

__device__ __forceinline__ f32x8 wmma_bf16(bf16x16 a, bf16x16 b, f32x8 c) {
  // v_wmma_f32_16x16x32_bf16: D = A(16x32) * B(32x16) + C(16x16 f32)
  return __builtin_amdgcn_wmma_f32_16x16x32_bf16(false, a, false, b, (short)0, c,
                                                 false, false);
}

// A-fragment from row-major [16 x K] bf16 buffer with leading dim ld.
// Lane layout (this file's k order, shared with load_B): M = lane%16; element e maps to
// K = half*16 + e  -> two 16B loads.
__device__ __forceinline__ bf16x16 load_A(const bf16* base, int ld, int k0, int lane) {
  const int m = lane & 15, half = lane >> 4;
  const bf16* p = base + (size_t)m * ld + k0 + half * 16;
  bf16x8 lo = *(const bf16x8*)(p);
  bf16x8 hi = *(const bf16x8*)(p + 8);
  bf16x16 r;
#pragma unroll
  for (int i = 0; i < 8; ++i) { r[i] = lo[i]; r[i + 8] = hi[i]; }
  return r;
}

// B-fragment for computing X @ W^T from row-major W [Nout x K]:
// B element (K,N) = W[n0+N][k0+K]; lane: N = lane%16, K = half*16 + e
// -> one contiguous 32B load per lane.
__device__ __forceinline__ bf16x16 load_B(const bf16* W, int ld, int n0, int k0, int lane) {
  const int n = lane & 15, half = lane >> 4;
  return *(const bf16x16*)(W + (size_t)(n0 + n) * ld + k0 + half * 16);
}

// D tile (16x16 f32) to row-major f32 buffer: N = lane%16, M = vgpr + 8*half.
__device__ __forceinline__ void store_D(float* out, int ld, int n0, int lane, f32x8 d) {
  const int n = lane & 15, mh = (lane >> 4) * 8;
#pragma unroll
  for (int i = 0; i < 8; ++i) out[(size_t)(mh + i) * ld + n0 + n] = d[i];
}

__device__ __forceinline__ float sigmoidf_(float x) { return 1.f / (1.f + expf(-x)); }

// ---------------------------------------------------------------------------
// Persistent scan kernel: 16 blocks, each owns 16 batch rows for all 50 steps.
// ---------------------------------------------------------------------------
__global__ __launch_bounds__(256) void kvmn_scan(
    const int* __restrict__ X, const float* __restrict__ E,
    const float* __restrict__ KBE, const float* __restrict__ r_matrix,
    const float* __restrict__ b_ih, const float* __restrict__ b_hh,
    const float* __restrict__ MNu_b, const float* __restrict__ MNu2_b,
    const float* __restrict__ MNea_b, const float* __restrict__ mlp1_b,
    const bf16* __restrict__ Wih, const bf16* __restrict__ Whh,
    const bf16* __restrict__ Wea, const bf16* __restrict__ Wu,
    const bf16* __restrict__ Wu2, const bf16* __restrict__ W1,
    float* __restrict__ MN,   // (B, NF, KBD) workspace, zeroed here
    bf16* __restrict__ yB,    // (B, HID) bf16 workspace out
    float* __restrict__ yOut) // (B, OUT) f32 -> d_out tail
{
  __shared__ float sh_h[16 * HID];     // 16 KB  (persistent h)
  __shared__ float sh_ysum[16 * OUT_]; // 16 KB  (persistent y accumulator)
  __shared__ bf16  sh_A[16 * 384];     // 12 KB  (A-staging: x / h / kb / [h|AC] / Ut)
  __shared__ float sh_gi[16 * G3];     // 48 KB  (gi; reused for Ut)
  __shared__ float sh_gh[16 * G3];     // 48 KB  (gh; reused for y_out)
  __shared__ float sh_t1[16 * KBD];    //  8 KB  (EA / U)
  __shared__ float sh_g[16 * 16];      //  1 KB  (gate / AW)
  __shared__ float sh_r[NF * KBD];     //  5 KB  (r_matrix cache)

  const int tid = threadIdx.x;
  const int lane = tid & 31, wave = tid >> 5;
  const int b0 = blockIdx.x * 16;
  float* MNblk = MN + (size_t)b0 * NF * KBD;

  for (int i = tid; i < 16 * HID; i += 256) { sh_h[i] = 0.f; sh_ysum[i] = 0.f; }
  for (int i = tid; i < 16 * NF * KBD; i += 256) MNblk[i] = 0.f;
  for (int i = tid; i < NF * KBD; i += 256) sh_r[i] = r_matrix[i];
  __syncthreads();

  for (int t = 0; t < T_; ++t) {
    // ---- stage x_t = E[X[b,t]] (row 0 masked) as bf16, ld=256 ----
    for (int i = tid; i < 16 * EMB; i += 256) {
      int m = i >> 8, k = i & 255;
      int idx = X[(b0 + m) * T_ + t];
      float v = (idx == 0) ? 0.f : E[(size_t)idx * EMB + k];
      sh_A[m * EMB + k] = (bf16)v;
    }
    __syncthreads();
    // ---- gi = x @ W_ih^T : 16x768, K=256 ----
    for (int nt = wave; nt < 48; nt += 8) {
      f32x8 acc = {};
#pragma unroll
      for (int kk = 0; kk < 8; ++kk) {
        bf16x16 a = load_A(sh_A, EMB, kk * 32, lane);
        bf16x16 b = load_B(Wih, EMB, nt * 16, kk * 32, lane);
        acc = wmma_bf16(a, b, acc);
      }
      store_D(sh_gi, G3, nt * 16, lane, acc);
    }
    __syncthreads();
    // ---- stage h as bf16, ld=256 ----
    for (int i = tid; i < 16 * HID; i += 256) sh_A[i] = (bf16)sh_h[i];
    __syncthreads();
    // ---- gh = h @ W_hh^T : 16x768, K=256 ----
    for (int nt = wave; nt < 48; nt += 8) {
      f32x8 acc = {};
#pragma unroll
      for (int kk = 0; kk < 8; ++kk) {
        bf16x16 a = load_A(sh_A, HID, kk * 32, lane);
        bf16x16 b = load_B(Whh, HID, nt * 16, kk * 32, lane);
        acc = wmma_bf16(a, b, acc);
      }
      store_D(sh_gh, G3, nt * 16, lane, acc);
    }
    __syncthreads();
    // ---- GRU elementwise -> sh_h ----
    for (int i = tid; i < 16 * HID; i += 256) {
      int m = i >> 8, j = i & 255;
      float ir = sh_gi[m * G3 + j] + b_ih[j];
      float iz = sh_gi[m * G3 + 256 + j] + b_ih[256 + j];
      float in = sh_gi[m * G3 + 512 + j] + b_ih[512 + j];
      float hr = sh_gh[m * G3 + j] + b_hh[j];
      float hz = sh_gh[m * G3 + 256 + j] + b_hh[256 + j];
      float hn = sh_gh[m * G3 + 512 + j] + b_hh[512 + j];
      float r = sigmoidf_(ir + hr);
      float z = sigmoidf_(iz + hz);
      float n = tanhf(in + r * hn);
      sh_h[i] = (1.f - z) * n + z * sh_h[i];
    }
    __syncthreads();
    // ---- stage kb_t = KBE[X[b,t]] (row 0 masked) bf16, ld=128 ----
    for (int i = tid; i < 16 * KBD; i += 256) {
      int m = i >> 7, k = i & 127;
      int idx = X[(b0 + m) * T_ + t];
      float v = (idx == 0) ? 0.f : KBE[(size_t)idx * KBD + k];
      sh_A[m * KBD + k] = (bf16)v;
    }
    __syncthreads();
    // ---- EA = kb @ MNea_W^T : 16x128, K=128 ----
    {
      int nt = wave;
      f32x8 acc = {};
#pragma unroll
      for (int kk = 0; kk < 4; ++kk) {
        bf16x16 a = load_A(sh_A, KBD, kk * 32, lane);
        bf16x16 b = load_B(Wea, KBD, nt * 16, kk * 32, lane);
        acc = wmma_bf16(a, b, acc);
      }
      store_D(sh_t1, KBD, nt * 16, lane, acc);
    }
    __syncthreads();
    for (int i = tid; i < 16 * KBD; i += 256)
      sh_t1[i] = tanhf(sh_t1[i] + MNea_b[i & 127]);
    __syncthreads();
    // ---- gate[m,f] = sigmoid(sum_d MN * (EA + r)) ----
    if (tid < 16 * NF) {
      int m = tid / NF, f = tid % NF;
      float s = 0.f;
      for (int d = 0; d < KBD; ++d)
        s += MNblk[(m * NF + f) * KBD + d] * (sh_t1[m * KBD + d] + sh_r[f * KBD + d]);
      sh_g[m * 16 + f] = sigmoidf_(s);
    }
    __syncthreads();
    // ---- MN = MN*(1-g) + EA2*g ----
    for (int i = tid; i < 16 * NF * KBD; i += 256) {
      int m = i / (NF * KBD);
      int rem = i - m * NF * KBD;
      int f = rem >> 7, d = rem & 127;
      float ea2 = sh_t1[m * KBD + d] + sh_r[f * KBD + d];
      float g = sh_g[m * 16 + f];
      MNblk[i] = MNblk[i] * (1.f - g) + ea2 * g;
    }
    __syncthreads();
    // ---- stage h bf16 into sh_A cols [0,256), ld=384 ----
    for (int i = tid; i < 16 * HID; i += 256) {
      int m = i >> 8, j = i & 255;
      sh_A[m * 384 + j] = (bf16)sh_h[i];
    }
    __syncthreads();
    // ---- U = tanh(h @ MNu_W^T + b) : 16x128, K=256 ----
    {
      int nt = wave;
      f32x8 acc = {};
#pragma unroll
      for (int kk = 0; kk < 8; ++kk) {
        bf16x16 a = load_A(sh_A, 384, kk * 32, lane);
        bf16x16 b = load_B(Wu, HID, nt * 16, kk * 32, lane);
        acc = wmma_bf16(a, b, acc);
      }
      store_D(sh_t1, KBD, nt * 16, lane, acc);
    }
    __syncthreads();
    for (int i = tid; i < 16 * KBD; i += 256)
      sh_t1[i] = tanhf(sh_t1[i] + MNu_b[i & 127]);
    __syncthreads();
    // ---- AW = softmax(U @ r^T) over NF ----
    if (tid < 16 * NF) {
      int m = tid / NF, f = tid % NF;
      float s = 0.f;
      for (int d = 0; d < KBD; ++d) s += sh_t1[m * KBD + d] * sh_r[f * KBD + d];
      sh_g[m * 16 + f] = s;
    }
    __syncthreads();
    if (tid < 16) {
      float mx = -1e30f;
      for (int f = 0; f < NF; ++f) mx = fmaxf(mx, sh_g[tid * 16 + f]);
      float s = 0.f;
      for (int f = 0; f < NF; ++f) {
        float e = expf(sh_g[tid * 16 + f] - mx);
        sh_g[tid * 16 + f] = e;
        s += e;
      }
      float inv = 1.f / s;
      for (int f = 0; f < NF; ++f) sh_g[tid * 16 + f] *= inv;
    }
    __syncthreads();
    // ---- AC = einsum(MN, AW) -> sh_A cols [256,384) bf16 ----
    for (int i = tid; i < 16 * KBD; i += 256) {
      int m = i >> 7, d = i & 127;
      float s = 0.f;
      for (int f = 0; f < NF; ++f)
        s += MNblk[(m * NF + f) * KBD + d] * sh_g[m * 16 + f];
      sh_A[m * 384 + 256 + d] = (bf16)s;
    }
    __syncthreads();
    // ---- Ut = tanh([h|AC] @ MNu2_W^T + b) : 16x256, K=384 -> sh_gi (ld 256) ----
    for (int nt = wave; nt < 16; nt += 8) {
      f32x8 acc = {};
#pragma unroll
      for (int kk = 0; kk < 12; ++kk) {
        bf16x16 a = load_A(sh_A, 384, kk * 32, lane);
        bf16x16 b = load_B(Wu2, 384, nt * 16, kk * 32, lane);
        acc = wmma_bf16(a, b, acc);
      }
      store_D(sh_gi, HID, nt * 16, lane, acc);
    }
    __syncthreads();
    for (int i = tid; i < 16 * HID; i += 256)
      sh_A[i] = (bf16)tanhf(sh_gi[i] + MNu2_b[i & 255]); // Ut bf16, ld=256
    __syncthreads();
    // ---- y_out = tanh(Ut @ mlp1_W^T + b); ysum += : 16x256, K=256 ----
    for (int nt = wave; nt < 16; nt += 8) {
      f32x8 acc = {};
#pragma unroll
      for (int kk = 0; kk < 8; ++kk) {
        bf16x16 a = load_A(sh_A, HID, kk * 32, lane);
        bf16x16 b = load_B(W1, HID, nt * 16, kk * 32, lane);
        acc = wmma_bf16(a, b, acc);
      }
      store_D(sh_gh, OUT_, nt * 16, lane, acc);
    }
    __syncthreads();
    for (int i = tid; i < 16 * OUT_; i += 256)
      sh_ysum[i] += tanhf(sh_gh[i] + mlp1_b[i & 255]);
    __syncthreads();
  } // t loop

  // ---- y = ysum / T ----
  for (int i = tid; i < 16 * OUT_; i += 256) {
    int m = i >> 8, j = i & 255;
    float v = sh_ysum[i] * (1.f / (float)T_);
    yOut[(size_t)(b0 + m) * OUT_ + j] = v;
    yB[(size_t)(b0 + m) * OUT_ + j] = (bf16)v;
  }
}

// ---------------------------------------------------------------------------
// Fused Sy + scores kernel: one block per 16-item tile. Sy never hits HBM.
// ---------------------------------------------------------------------------
__global__ __launch_bounds__(256) void kvmn_scores(
    const float* __restrict__ E, const float* __restrict__ KBE,
    const float* __restrict__ mlp2_b, const bf16* __restrict__ W2,
    const bf16* __restrict__ yB, float* __restrict__ scores)
{
  __shared__ bf16  sh_Ain[16 * 384];
  __shared__ float sh_Sy[16 * 256];
  __shared__ bf16  sh_SyB[16 * 256];
  const int tid = threadIdx.x, lane = tid & 31, wave = tid >> 5;
  const int n0 = blockIdx.x * 16;

  // stage MergeE tile = [E | KBE] rows n0..n0+15 (row 0 masked) as bf16
  for (int i = tid; i < 16 * 384; i += 256) {
    int r = i / 384, k = i - r * 384;
    int item = n0 + r;
    float v = 0.f;
    if (item != 0)
      v = (k < 256) ? E[(size_t)item * EMB + k] : KBE[(size_t)item * KBD + (k - 256)];
    sh_Ain[i] = (bf16)v;
  }
  __syncthreads();
  // Sy tile = MergeE_tile @ mlp2_W^T : 16x256, K=384
  for (int nt = wave; nt < 16; nt += 8) {
    f32x8 acc = {};
#pragma unroll
    for (int kk = 0; kk < 12; ++kk) {
      bf16x16 a = load_A(sh_Ain, 384, kk * 32, lane);
      bf16x16 b = load_B(W2, 384, nt * 16, kk * 32, lane);
      acc = wmma_bf16(a, b, acc);
    }
    store_D(sh_Sy, 256, nt * 16, lane, acc);
  }
  __syncthreads();
  for (int i = tid; i < 16 * 256; i += 256)
    sh_SyB[i] = (bf16)tanhf(sh_Sy[i] + mlp2_b[i & 255]);
  __syncthreads();
  // scores[:, n0:n0+16] = y @ SyTile^T : M=256 (batch), N=16, K=256
  for (int mt = wave; mt < 16; mt += 8) {
    f32x8 acc = {};
#pragma unroll
    for (int kk = 0; kk < 8; ++kk) {
      bf16x16 a = load_A(yB + (size_t)mt * 16 * 256, 256, kk * 32, lane);
      bf16x16 b = load_B(sh_SyB, 256, 0, kk * 32, lane);
      acc = wmma_bf16(a, b, acc);
    }
    const int n = lane & 15, mh = (lane >> 4) * 8;
#pragma unroll
    for (int i = 0; i < 8; ++i)
      scores[(size_t)(mt * 16 + mh + i) * NITEMS + n0 + n] = acc[i];
  }
}

// ---------------------------------------------------------------------------
// Row-wise log_softmax over N=100000, in place. One block per batch row.
// ---------------------------------------------------------------------------
__global__ __launch_bounds__(256) void kvmn_logsoftmax(float* __restrict__ scores)
{
  __shared__ float red[256];
  const int tid = threadIdx.x;
  float* p = scores + (size_t)blockIdx.x * NITEMS;

  float mx = -3.4e38f;
  for (int i = tid; i < NITEMS; i += 256) mx = fmaxf(mx, p[i]);
  red[tid] = mx;
  __syncthreads();
  for (int s = 128; s > 0; s >>= 1) {
    if (tid < s) red[tid] = fmaxf(red[tid], red[tid + s]);
    __syncthreads();
  }
  mx = red[0];
  __syncthreads();

  float sum = 0.f;
  for (int i = tid; i < NITEMS; i += 256) sum += expf(p[i] - mx);
  red[tid] = sum;
  __syncthreads();
  for (int s = 128; s > 0; s >>= 1) {
    if (tid < s) red[tid] += red[tid + s];
    __syncthreads();
  }
  const float lse = logf(red[0]) + mx;
  __syncthreads();

  for (int i = tid; i < NITEMS; i += 256) p[i] -= lse;
}

__global__ void f32_to_bf16(const float* __restrict__ src, bf16* __restrict__ dst, int n) {
  int i = blockIdx.x * blockDim.x + threadIdx.x;
  if (i < n) dst[i] = (bf16)src[i];
}

// ---------------------------------------------------------------------------
extern "C" void kernel_launch(void* const* d_in, const int* in_sizes, int n_in,
                              void* d_out, int out_size, void* d_ws, size_t ws_size,
                              hipStream_t stream) {
  (void)in_sizes; (void)n_in; (void)out_size; (void)ws_size;
  const int*   X      = (const int*)d_in[0];
  const float* E      = (const float*)d_in[1];
  const float* KBE    = (const float*)d_in[2];
  const float* r_mat  = (const float*)d_in[3];
  const float* W_ih   = (const float*)d_in[4];
  const float* W_hh   = (const float*)d_in[5];
  const float* b_ih   = (const float*)d_in[6];
  const float* b_hh   = (const float*)d_in[7];
  const float* MNu_W  = (const float*)d_in[8];
  const float* MNu_b  = (const float*)d_in[9];
  const float* MNu2_W = (const float*)d_in[10];
  const float* MNu2_b = (const float*)d_in[11];
  const float* MNea_W = (const float*)d_in[12];
  const float* MNea_b = (const float*)d_in[13];
  const float* mlp1_W = (const float*)d_in[14];
  const float* mlp1_b = (const float*)d_in[15];
  const float* mlp2_W = (const float*)d_in[16];
  const float* mlp2_b = (const float*)d_in[17];

  char* ws = (char*)d_ws;
  size_t off = 0;
  auto alloc = [&](size_t bytes) -> void* {
    void* p = ws + off;
    off = (off + bytes + 255) & ~(size_t)255;
    return p;
  };
  bf16* WihB = (bf16*)alloc((size_t)G3 * EMB * 2);
  bf16* WhhB = (bf16*)alloc((size_t)G3 * HID * 2);
  bf16* WeaB = (bf16*)alloc((size_t)KBD * KBD * 2);
  bf16* WuB  = (bf16*)alloc((size_t)KBD * HID * 2);
  bf16* Wu2B = (bf16*)alloc((size_t)HID * 384 * 2);
  bf16* W1B  = (bf16*)alloc((size_t)OUT_ * HID * 2);
  bf16* W2B  = (bf16*)alloc((size_t)OUT_ * 384 * 2);
  bf16* yB   = (bf16*)alloc((size_t)B_ * HID * 2);
  float* MNw = (float*)alloc((size_t)B_ * NF * KBD * 4);

  auto conv = [&](const float* s, bf16* d, int n) {
    f32_to_bf16<<<(n + 255) / 256, 256, 0, stream>>>(s, d, n);
  };
  conv(W_ih, WihB, G3 * EMB);
  conv(W_hh, WhhB, G3 * HID);
  conv(MNea_W, WeaB, KBD * KBD);
  conv(MNu_W, WuB, KBD * HID);
  conv(MNu2_W, Wu2B, HID * 384);
  conv(mlp1_W, W1B, OUT_ * HID);
  conv(mlp2_W, W2B, OUT_ * 384);

  float* scores = (float*)d_out;
  float* yOut = scores + (size_t)B_ * NITEMS;

  kvmn_scan<<<B_ / 16, 256, 0, stream>>>(
      X, E, KBE, r_mat, b_ih, b_hh, MNu_b, MNu2_b, MNea_b, mlp1_b,
      WihB, WhhB, WeaB, WuB, Wu2B, W1B, MNw, yB, yOut);

  kvmn_scores<<<NITEMS / 16, 256, 0, stream>>>(E, KBE, mlp2_b, W2B, yB, scores);

  kvmn_logsoftmax<<<B_, 256, 0, stream>>>(scores);
}
